// HungarianMatcher_31396210934228
// MI455X (gfx1250) — compile-verified, matrix operand order checked by reading, not run on hardware
//
#include <hip/hip_runtime.h>
#include <hip/hip_bf16.h>

// Problem constants (match reference)
#define BN 16
#define NN 1000
#define MM 100
#define CC 80
#define PP 1024          // T*T = 32*32
#define NTILES 63        // ceil(1000/16)
#define MTILES 7         // ceil(100/16)
#define EPSF 1e-8f
#define BIGF 1e8f

typedef __attribute__((ext_vector_type(8)))  _Float16 v8h;
typedef __attribute__((ext_vector_type(16))) _Float16 v16h;
typedef __attribute__((ext_vector_type(8)))  float    v8f;

// ---------------------------------------------------------------------------
// Pass 1: pm = sigmoid(pred_masks) -> f16 staging + per-row sum
// one block per (b,n) row of 1024 elements
// ---------------------------------------------------------------------------
__global__ void pm_sigmoid_kernel(const float* __restrict__ src,
                                  _Float16* __restrict__ dst,
                                  float* __restrict__ rowsum) {
    const int row = blockIdx.x;                       // b*NN + n
    const float* s = src + (size_t)row * PP;
    _Float16*    d = dst + (size_t)row * PP;
    float acc = 0.f;
    for (int i = threadIdx.x; i < PP; i += blockDim.x) {
        float p = 1.0f / (1.0f + __expf(-s[i]));
        d[i] = (_Float16)p;
        acc += p;
    }
    for (int off = 16; off > 0; off >>= 1) acc += __shfl_down(acc, off, 32);
    __shared__ float red[8];
    if ((threadIdx.x & 31) == 0) red[threadIdx.x >> 5] = acc;
    __syncthreads();
    if (threadIdx.x == 0) {
        float t = 0.f;
        #pragma unroll
        for (int i = 0; i < 8; ++i) t += red[i];
        rowsum[row] = t;
    }
}

// ---------------------------------------------------------------------------
// Pass 2: gt_masks (already 0/1 float) -> f16 staging + per-row sum
// ---------------------------------------------------------------------------
__global__ void gm_convert_kernel(const float* __restrict__ src,
                                  _Float16* __restrict__ dst,
                                  float* __restrict__ rowsum) {
    const int row = blockIdx.x;                       // b*MM + m
    const float* s = src + (size_t)row * PP;
    _Float16*    d = dst + (size_t)row * PP;
    float acc = 0.f;
    for (int i = threadIdx.x; i < PP; i += blockDim.x) {
        float g = s[i];
        d[i] = (_Float16)g;
        acc += g;
    }
    for (int off = 16; off > 0; off >>= 1) acc += __shfl_down(acc, off, 32);
    __shared__ float red[8];
    if ((threadIdx.x & 31) == 0) red[threadIdx.x >> 5] = acc;
    __syncthreads();
    if (threadIdx.x == 0) {
        float t = 0.f;
        #pragma unroll
        for (int i = 0; i < 8; ++i) t += red[i];
        rowsum[row] = t;
    }
}

// ---------------------------------------------------------------------------
// Main: one wave per 16x16 (n,m) tile. K=1024 dice-intersection GEMM via
// v_wmma_f32_16x16x32_f16, fused epilogue computes the full matching cost.
// ---------------------------------------------------------------------------
__global__ void __launch_bounds__(256, 2)
matcher_wmma_kernel(const _Float16* __restrict__ pmh,
                    const _Float16* __restrict__ gmh,
                    const float* __restrict__ pm_sum,
                    const float* __restrict__ gm_sum,
                    const float* __restrict__ pred_logits,
                    const float* __restrict__ pred_boxes,
                    const float* __restrict__ gt_boxes,
                    const int*   __restrict__ gt_classes,
                    const float* __restrict__ image_size,
                    float* __restrict__ out) {
    const int lane = threadIdx.x & 31;
    const int wave = blockIdx.x * (blockDim.x >> 5) + (threadIdx.x >> 5);
    const int mt = wave % MTILES;
    const int t1 = wave / MTILES;
    const int nt = t1 % NTILES;
    const int b  = t1 / NTILES;
    if (b >= BN) return;                 // wave-uniform guard

    const int halfsel = lane >> 4;       // 0: lanes 0-15, 1: lanes 16-31
    const int kbase   = halfsel * 8;     // K phase per ISA 16-bit A layout
    const int rlane   = lane & 15;

    // Clamp edge-tile rows so every lane issues valid loads (EXEC all-ones
    // is required for WMMA); clamped duplicates are skipped at store time.
    int n_row = nt * 16 + rlane; if (n_row > NN - 1) n_row = NN - 1;
    int m_row = mt * 16 + rlane; if (m_row > MM - 1) m_row = MM - 1;

    const _Float16* arow = pmh + ((size_t)(b * NN + n_row)) * PP + kbase;
    const _Float16* brow = gmh + ((size_t)(b * MM + m_row)) * PP + kbase;

    v8f acc = {};
    #pragma unroll 4
    for (int k0 = 0; k0 < PP; k0 += 32) {
        // Fragment layout (16-bit A 16x32): lane holds K {kbase..kbase+7}
        // and {kbase+16..kbase+23} -> two contiguous 16B chunks.
        v8h a0 = *(const v8h*)(arow + k0);
        v8h a1 = *(const v8h*)(arow + k0 + 16);
        v8h b0 = *(const v8h*)(brow + k0);
        v8h b1 = *(const v8h*)(brow + k0 + 16);
        v16h av = __builtin_shufflevector(a0, a1, 0,1,2,3,4,5,6,7,8,9,10,11,12,13,14,15);
        v16h bv = __builtin_shufflevector(b0, b1, 0,1,2,3,4,5,6,7,8,9,10,11,12,13,14,15);
        acc = __builtin_amdgcn_wmma_f32_16x16x32_f16(
                  /*neg_a=*/false, av, /*neg_b=*/false, bv,
                  /*c_mod=*/(short)0, acc, /*reuse_a=*/false, /*reuse_b=*/false);
    }

    // ---- epilogue: fuse cls / L1 / GIoU / in-box with dice ----
    const int m = mt * 16 + rlane;       // C/D layout: column = lane % 16
    if (m >= MM) return;

    const float4 gb  = ((const float4*)gt_boxes)[b * MM + m];
    const int    cls = gt_classes[b * MM + m];
    const float  gms = gm_sum[b * MM + m];
    const float4 img = ((const float4*)image_size)[b];
    const float gx0 = gb.x / img.x, gy0 = gb.y / img.y;
    const float gx1 = gb.z / img.z, gy1 = gb.w / img.w;
    const float garea = (gx1 - gx0) * (gy1 - gy0);

    #pragma unroll
    for (int v = 0; v < 8; ++v) {
        // C/D layout: VGPR v holds row M = v (lanes 0-15) / v+8 (lanes 16-31)
        const int n = nt * 16 + v + halfsel * 8;
        if (n >= NN) continue;

        // mask dice cost
        const float inter_m = acc[v];
        const float denom = pm_sum[b * NN + n] + gms + EPSF;
        const float mask_cost = 1.0f - 2.0f * inter_m / denom;

        // focal classification cost, gathered at gt class
        const float x = pred_logits[((size_t)(b * NN + n)) * CC + cls];
        const float p = 1.0f / (1.0f + __expf(-x));
        const float pos = 0.25f * (1.0f - p) * (1.0f - p) * (-__logf(p + EPSF));
        const float neg = 0.75f * p * p * (-__logf(1.0f - p + EPSF));
        const float cls_cost = pos - neg;

        const float4 pb = ((const float4*)pred_boxes)[b * NN + n];
        const float px0 = pb.x / img.x, py0 = pb.y / img.y;
        const float px1 = pb.z / img.z, py1 = pb.w / img.w;

        // L1 (normalized)
        const float l1 = fabsf(px0 - gx0) + fabsf(py0 - gy0)
                       + fabsf(px1 - gx1) + fabsf(py1 - gy1);

        // GIoU (normalized)
        const float parea = (px1 - px0) * (py1 - py0);
        const float iw = fmaxf(fminf(px1, gx1) - fmaxf(px0, gx0), 0.f);
        const float ih = fmaxf(fminf(py1, gy1) - fmaxf(py0, gy0), 0.f);
        const float ib = iw * ih;
        const float un = parea + garea - ib;
        const float iou = ib / un;
        const float cw = fmaxf(px1, gx1) - fminf(px0, gx0);
        const float ch = fmaxf(py1, gy1) - fminf(py0, gy0);
        const float ca = cw * ch;
        const float giou_cost = 1.0f - (iou - (ca - un) / ca);

        float cost = 2.0f * cls_cost + 5.0f * l1
                   + 2.0f * giou_cost + 5.0f * mask_cost;

        // out-of-box penalty (raw coords, strict inequalities)
        const float pcx = 0.5f * (pb.x + pb.z);
        const float pcy = 0.5f * (pb.y + pb.w);
        const bool inb = (pcx > gb.x) && (pcx < gb.z) &&
                         (pcy > gb.y) && (pcy < gb.w);
        if (!inb) cost += BIGF;

        out[((size_t)(b * NN + n)) * MM + m] = cost;
    }
}

// ---------------------------------------------------------------------------
extern "C" void kernel_launch(void* const* d_in, const int* in_sizes, int n_in,
                              void* d_out, int out_size, void* d_ws, size_t ws_size,
                              hipStream_t stream) {
    const float* pred_logits = (const float*)d_in[0];
    const float* pred_boxes  = (const float*)d_in[1];
    const float* pred_masks  = (const float*)d_in[2];
    const float* gt_boxes    = (const float*)d_in[3];
    const float* gt_masks    = (const float*)d_in[4];
    const int*   gt_classes  = (const int*)d_in[5];
    const float* image_size  = (const float*)d_in[6];
    float* out = (float*)d_out;

    // workspace layout (16B-aligned slices)
    char* ws = (char*)d_ws;
    const size_t pmh_bytes = (size_t)BN * NN * PP * sizeof(_Float16); // 32,768,000
    const size_t gmh_bytes = (size_t)BN * MM * PP * sizeof(_Float16); //  3,276,800
    const size_t pms_bytes = (size_t)BN * NN * sizeof(float);         //     64,000
    _Float16* pmh    = (_Float16*)(ws);
    _Float16* gmh    = (_Float16*)(ws + pmh_bytes);
    float*    pm_sum = (float*)(ws + pmh_bytes + gmh_bytes);
    float*    gm_sum = (float*)(ws + pmh_bytes + gmh_bytes + pms_bytes);

    pm_sigmoid_kernel<<<BN * NN, 256, 0, stream>>>(pred_masks, pmh, pm_sum);
    gm_convert_kernel<<<BN * MM, 256, 0, stream>>>(gt_masks, gmh, gm_sum);

    const int total_waves = BN * NTILES * MTILES;   // 7056
    const int waves_per_block = 8;                  // 256 threads, wave32
    const int blocks = (total_waves + waves_per_block - 1) / waves_per_block; // 882
    matcher_wmma_kernel<<<blocks, 256, 0, stream>>>(
        pmh, gmh, pm_sum, gm_sum,
        pred_logits, pred_boxes, gt_boxes, gt_classes, image_size, out);
}
